// TripletEntropyLoss_19009525252700
// MI455X (gfx1250) — matrix operand order repaired and checked by reading, not compiled, not run on hardware
//
#include <hip/hip_runtime.h>

typedef __attribute__((ext_vector_type(16))) _Float16 v16h;
typedef __attribute__((ext_vector_type(8)))  _Float16 v8h;
typedef __attribute__((ext_vector_type(4)))  _Float16 v4h;
typedef __attribute__((ext_vector_type(8)))  float    v8f;

#define NROWS   8192
#define DDIM    128
#define DELTA_C 0.001f
// exp(sim/4)*coeff == exp2(sim * LOG2E_DIV_MAG + log2(coeff))
#define LOG2E_DIV_MAG 0.3606737602f   /* log2(e)/4 */

// ---------------------------------------------------------------------------
// Kernel 1: split fp32 -> f16 hi + f16 lo (Markidis split for ~fp32 accuracy)
// ---------------------------------------------------------------------------
__global__ void convert_split(const float* __restrict__ src,
                              _Float16* __restrict__ hi,
                              _Float16* __restrict__ lo,
                              int n4 /* elements/4 */) {
    int i = blockIdx.x * blockDim.x + threadIdx.x;
    if (i >= n4) return;
    float4 v = ((const float4*)src)[i];
    float x[4] = {v.x, v.y, v.z, v.w};
    v4h h, l;
#pragma unroll
    for (int j = 0; j < 4; ++j) {
        _Float16 hh = (_Float16)x[j];
        h[j] = hh;
        l[j] = (_Float16)(x[j] - (float)hh);
    }
    ((v4h*)hi)[i] = h;
    ((v4h*)lo)[i] = l;
}

// ---------------------------------------------------------------------------
// Kernel 2: pos_sim[i] = (a_i . p_i) / MAG in full fp32; zero rowSum
// ---------------------------------------------------------------------------
__global__ void possim_rowinit(const float* __restrict__ a,
                               const float* __restrict__ p,
                               float* __restrict__ posSim,
                               float* __restrict__ rowSum) {
    int wave = threadIdx.x >> 5;
    int lane = threadIdx.x & 31;
    int row  = blockIdx.x * 8 + wave;
    const float4 va = ((const float4*)(a + (size_t)row * DDIM))[lane];
    const float4 vp = ((const float4*)(p + (size_t)row * DDIM))[lane];
    float s = va.x * vp.x + va.y * vp.y + va.z * vp.z + va.w * vp.w;
    s += __shfl_xor(s, 1);
    s += __shfl_xor(s, 2);
    s += __shfl_xor(s, 4);
    s += __shfl_xor(s, 8);
    s += __shfl_xor(s, 16);
    if (lane == 0) {
        posSim[row] = s * 0.25f;
        rowSum[row] = 0.0f;
    }
}

// ---------------------------------------------------------------------------
// One 16x16 output tile against K=128. Scheduling intent (pinned with
// sched_group_barrier): software pipeline — loads(ks0,ks1), wmma(ks0),
// loads(ks2), wmma(ks1), loads(ks3), wmma(ks2,ks3) — so the first WMMA only
// needs the first 4 loads and the rest retire under the matrix pipe.
// ---------------------------------------------------------------------------
template <bool DIAG>
__device__ __forceinline__ void do_tile(
        const _Float16* __restrict__ Bh, const _Float16* __restrict__ Bl,
        float bias, int colBase, int rowBase, int hsel, int lidx,
        const v16h (&afh)[4], const v16h (&afl)[4], float (&rowAcc)[8]) {
    // B 16-bit 32x16 layout: lane<16 -> N=lane, K=kb+{0-7, 16-23};
    //                        lane>=16 -> N=lane-16, K=kb+{8-15, 24-31}
    const size_t browOff = (size_t)(colBase + lidx) * DDIM + 8 * hsel;

    v16h bh[4], bl[4];
#pragma unroll
    for (int ks = 0; ks < 4; ++ks) {
        const _Float16* bp = Bh + browOff + ks * 32;
        const _Float16* bq = Bl + browOff + ks * 32;
        v8h b0 = *(const v8h*)bp, b1 = *(const v8h*)(bp + 16);
        v8h c0 = *(const v8h*)bq, c1 = *(const v8h*)(bq + 16);
#pragma unroll
        for (int i = 0; i < 8; ++i) {
            bh[ks][i] = b0[i]; bh[ks][i + 8] = b1[i];
            bl[ks][i] = c0[i]; bl[ks][i + 8] = c1[i];
        }
    }

    // Three independent chains (hi*hi, hi*lo, lo*hi): adjacent WMMAs have no
    // RAW dependency, so the matrix pipe issues back-to-back.
    v8f chh = {}, chl = {}, clh = {};
#pragma unroll
    for (int ks = 0; ks < 4; ++ks) {
        chh = __builtin_amdgcn_wmma_f32_16x16x32_f16(false, afh[ks], false, bh[ks],
                                                     (short)0, chh, false, false);
        chl = __builtin_amdgcn_wmma_f32_16x16x32_f16(false, afh[ks], false, bl[ks],
                                                     (short)0, chl, false, false);
        clh = __builtin_amdgcn_wmma_f32_16x16x32_f16(false, afl[ks], false, bh[ks],
                                                     (short)0, clh, false, false);
    }

    // Pinned software pipeline: interleave VMEM-read groups with WMMA groups.
    __builtin_amdgcn_sched_group_barrier(0x020, 8, 0);  // loads ks0 + ks1
    __builtin_amdgcn_sched_group_barrier(0x008, 3, 0);  // wmma ks0
    __builtin_amdgcn_sched_group_barrier(0x020, 4, 0);  // loads ks2
    __builtin_amdgcn_sched_group_barrier(0x008, 3, 0);  // wmma ks1
    __builtin_amdgcn_sched_group_barrier(0x020, 4, 0);  // loads ks3
    __builtin_amdgcn_sched_group_barrier(0x008, 6, 0);  // wmma ks2 + ks3

    // C layout — VGPR r: lanes 0-15 M=r, lanes 16-31 M=r+8; N=lidx
    // coeff*exp(sim/4) == exp2(fma(sim, log2e/4, log2(coeff)))
    // arg stays in [-12,12] -> raw v_exp_f32, no libm subnormal fixup needed
    const int gcol = colBase + lidx;
#pragma unroll
    for (int r = 0; r < 8; ++r) {
        float sim = chh[r] + chl[r] + clh[r];
        if (DIAG && (rowBase + r + 8 * hsel == gcol)) sim = DELTA_C;
        rowAcc[r] += __builtin_amdgcn_exp2f(fmaf(sim, LOG2E_DIV_MAG, bias));
    }
}

// ---------------------------------------------------------------------------
// Kernel 3: fused GEMM(x3) -> exp -> row-sum
// grid = (64 row-blocks, 16 col-slices), block = 256 (8 waves)
// wave w owns rows blockIdx.x*128 + w*16 .. +15; three specialized passes
// (negatives / positives / anchors) over 512 16-col tiles each.
// Barrier at loop TOP so it never lands between a tile's loads and its WMMAs.
// ---------------------------------------------------------------------------
__global__ __launch_bounds__(256) void fused_simexp(
        const _Float16* __restrict__ Ahi, const _Float16* __restrict__ Alo,
        const _Float16* __restrict__ Phi, const _Float16* __restrict__ Plo,
        const _Float16* __restrict__ Nhi, const _Float16* __restrict__ Nlo,
        float* __restrict__ rowSum) {
    const int lane = threadIdx.x & 31;
    const int wave = threadIdx.x >> 5;
    const int hsel = lane >> 4;      // 0: lanes 0-15, 1: lanes 16-31
    const int lidx = lane & 15;
    const int rowBase = blockIdx.x * 128 + wave * 16;
    const int myrow   = rowBase + lidx;

    // A fragments, resident across all three passes.
    // A 16-bit 16x32 layout: lane<16 -> M=lane, halves {K 0-7, K 16-23};
    //                        lane>=16 -> M=lane-16, halves {K 8-15, K 24-31}
    v16h afh[4], afl[4];
#pragma unroll
    for (int ks = 0; ks < 4; ++ks) {
        const int kb = ks * 32 + 8 * hsel;
        const _Float16* rh = Ahi + (size_t)myrow * DDIM + kb;
        const _Float16* rl = Alo + (size_t)myrow * DDIM + kb;
        v8h h0 = *(const v8h*)rh, h1 = *(const v8h*)(rh + 16);
        v8h l0 = *(const v8h*)rl, l1 = *(const v8h*)(rl + 16);
#pragma unroll
        for (int i = 0; i < 8; ++i) {
            afh[ks][i] = h0[i]; afh[ks][i + 8] = h1[i];
            afl[ks][i] = l0[i]; afl[ks][i + 8] = l1[i];
        }
    }

    float rowAcc[8] = {0.f, 0.f, 0.f, 0.f, 0.f, 0.f, 0.f, 0.f};

    // Pass 1: negatives (coeff 0.5 -> bias -1, no diagonal handling)
    for (int t = blockIdx.y; t < 512; t += 16) {
        __syncthreads();   // keep 8 waves on the same B tile (WGP$ reuse)
        do_tile<false>(Nhi, Nlo, -1.0f, t << 4, rowBase, hsel, lidx, afh, afl, rowAcc);
    }
    // Pass 2: positives (coeff 0.25 -> bias -2, diag -> DELTA)
    for (int t = blockIdx.y; t < 512; t += 16) {
        __syncthreads();
        do_tile<true>(Phi, Plo, -2.0f, t << 4, rowBase, hsel, lidx, afh, afl, rowAcc);
    }
    // Pass 3: anchors (coeff 0.25 -> bias -2, diag -> DELTA)
    for (int t = blockIdx.y; t < 512; t += 16) {
        __syncthreads();
        do_tile<true>(Ahi, Alo, -2.0f, t << 4, rowBase, hsel, lidx, afh, afl, rowAcc);
    }

    // cross-lane reduction within each half-wave (lanes sharing the same rows)
#pragma unroll
    for (int r = 0; r < 8; ++r) {
        float v = rowAcc[r];
        v += __shfl_xor(v, 1);
        v += __shfl_xor(v, 2);
        v += __shfl_xor(v, 4);
        v += __shfl_xor(v, 8);
        rowAcc[r] = v;
    }
    if (lidx == 0) {
#pragma unroll
        for (int r = 0; r < 8; ++r)
            atomicAdd(&rowSum[rowBase + r + 8 * hsel], rowAcc[r]);
    }
}

// ---------------------------------------------------------------------------
// Kernel 4: loss = mean(relu(log(rowSum) - posSim))
// ---------------------------------------------------------------------------
__global__ void finalize(const float* __restrict__ rowSum,
                         const float* __restrict__ posSim,
                         float* __restrict__ out) {
    __shared__ float sdata[8];
    float s = 0.f;
    for (int i = threadIdx.x; i < NROWS; i += 256) {
        float v = logf(rowSum[i]) - posSim[i];
        s += fmaxf(v, 0.f);
    }
    s += __shfl_xor(s, 1);
    s += __shfl_xor(s, 2);
    s += __shfl_xor(s, 4);
    s += __shfl_xor(s, 8);
    s += __shfl_xor(s, 16);
    if ((threadIdx.x & 31) == 0) sdata[threadIdx.x >> 5] = s;
    __syncthreads();
    if (threadIdx.x == 0) {
        float t = 0.f;
#pragma unroll
        for (int i = 0; i < 8; ++i) t += sdata[i];
        out[0] = t / (float)NROWS;
    }
}

// ---------------------------------------------------------------------------
extern "C" void kernel_launch(void* const* d_in, const int* in_sizes, int n_in,
                              void* d_out, int out_size, void* d_ws, size_t ws_size,
                              hipStream_t stream) {
    const float* a = (const float*)d_in[0];
    const float* p = (const float*)d_in[1];
    const float* n = (const float*)d_in[2];

    const size_t MAT = (size_t)NROWS * DDIM;          // 1,048,576 halfs per matrix
    char* ws = (char*)d_ws;
    _Float16* Ahi = (_Float16*)(ws + 0 * MAT * 2);
    _Float16* Alo = (_Float16*)(ws + 1 * MAT * 2);
    _Float16* Phi = (_Float16*)(ws + 2 * MAT * 2);
    _Float16* Plo = (_Float16*)(ws + 3 * MAT * 2);
    _Float16* Nhi = (_Float16*)(ws + 4 * MAT * 2);
    _Float16* Nlo = (_Float16*)(ws + 5 * MAT * 2);
    float* rowSum = (float*)(ws + 6 * MAT * 2);
    float* posSim = rowSum + NROWS;

    const int n4 = (int)(MAT / 4);                    // 262,144 threads per matrix
    convert_split<<<n4 / 256, 256, 0, stream>>>(a, Ahi, Alo, n4);
    convert_split<<<n4 / 256, 256, 0, stream>>>(p, Phi, Plo, n4);
    convert_split<<<n4 / 256, 256, 0, stream>>>(n, Nhi, Nlo, n4);

    possim_rowinit<<<NROWS / 8, 256, 0, stream>>>(a, p, posSim, rowSum);

    fused_simexp<<<dim3(64, 16), 256, 0, stream>>>(Ahi, Alo, Phi, Plo, Nhi, Nlo,
                                                   rowSum);

    finalize<<<1, 256, 0, stream>>>(rowSum, posSim, (float*)d_out);
}
